// EncoderLayer_69896297775108
// MI455X (gfx1250) — matrix operand order repaired and checked by reading, not compiled, not run on hardware
//
#include <hip/hip_runtime.h>
#include <hip/hip_bf16.h>
#include <math.h>

// ---------------- problem dims ----------------
#define S_LEN 2048
#define B_SZ  2
#define E_DIM 1024
#define H_NUM 16
#define F_DIM 4096
#define DH    64
#define NT    (S_LEN * B_SZ)   // 4096 rows of activations

// ---------------- vector types ----------------
typedef __attribute__((ext_vector_type(16))) _Float16 v16h;
typedef __attribute__((ext_vector_type(8)))  _Float16 h8;
typedef __attribute__((ext_vector_type(8)))  float    v8f;
typedef int i32x4 __attribute__((vector_size(16)));   // matches builtin's int4 pointee

union Frag {
    v16h v;
    struct { h8 lo, hi; };
    __device__ Frag() {}
};

__device__ inline v8f vzero() {
    v8f z;
#pragma unroll
    for (int i = 0; i < 8; ++i) z[i] = 0.f;
    return z;
}

__device__ inline v8f wmma16(const Frag& a, const Frag& b, v8f c) {
    // D = A(16x32 f16) * B(32x16 f16) + C(16x16 f32)
    return __builtin_amdgcn_wmma_f32_16x16x32_f16(
        /*neg_a=*/false, a.v, /*neg_b=*/false, b.v,
        /*c_mod=*/(short)0, c, /*reuse_a=*/false, /*reuse_b=*/false);
}

// -------- CDNA5 async global->LDS copy (ASYNCcnt path), builtin/asm hedge --------
__device__ inline void async_copy_b128(void* lds, const void* g) {
#if __has_builtin(__builtin_amdgcn_global_load_async_to_lds_b128)
    __builtin_amdgcn_global_load_async_to_lds_b128(
        (__attribute__((address_space(1))) i32x4*)g,
        (__attribute__((address_space(3))) i32x4*)lds,
        0, 0);
#elif defined(__AMDGCN__)
    asm volatile("global_load_async_to_lds_b128 %0, %1, off"
                 :: "v"((unsigned)(unsigned long long)lds),
                    "v"((unsigned long long)g)
                 : "memory");
#endif
}

__device__ inline void wait_async0() {
#if __has_builtin(__builtin_amdgcn_s_wait_asynccnt)
    __builtin_amdgcn_s_wait_asynccnt(0);
#elif defined(__AMDGCN__)
    asm volatile("s_wait_asynccnt 0x0" ::: "memory");
#endif
}

// ---------------- small utility kernels ----------------
__global__ void cvt_f16_kernel(const float* __restrict__ in,
                               _Float16* __restrict__ out, int n) {
    int i = blockIdx.x * blockDim.x + threadIdx.x;
    int stride = gridDim.x * blockDim.x;
    for (; i < n; i += stride) out[i] = (_Float16)in[i];
}

// mask: True (nonzero byte) = padded key -> additive bias -1e9
__global__ void maskbias_kernel(const unsigned char* __restrict__ m,
                                float* __restrict__ out, int n) {
    int i = blockIdx.x * blockDim.x + threadIdx.x;
    if (i < n) out[i] = m[i] ? -1.0e9f : 0.0f;
}

// LayerNorm over E=1024, one block (256 thr) per row, fp16 output
__global__ __launch_bounds__(256)
void ln_f16_kernel(const float* __restrict__ x, const float* __restrict__ g,
                   const float* __restrict__ bb, _Float16* __restrict__ out) {
    const int row = blockIdx.x;
    const int tid = threadIdx.x;
    const float* xr = x + (size_t)row * E_DIM;
    float4 v = ((const float4*)xr)[tid];
    float vv[4] = {v.x, v.y, v.z, v.w};
    float s = vv[0] + vv[1] + vv[2] + vv[3];
    float s2 = vv[0]*vv[0] + vv[1]*vv[1] + vv[2]*vv[2] + vv[3]*vv[3];
#pragma unroll
    for (int o = 1; o < 32; o <<= 1) {
        s  += __shfl_xor(s, o);
        s2 += __shfl_xor(s2, o);
    }
    __shared__ float sh[16];
    int w = tid >> 5, lane = tid & 31;
    if (lane == 0) { sh[w] = s; sh[8 + w] = s2; }
    __syncthreads();
    if (tid == 0) {
        float ts = 0.f, ts2 = 0.f;
#pragma unroll
        for (int i = 0; i < 8; ++i) { ts += sh[i]; ts2 += sh[8 + i]; }
        float mean = ts * (1.0f / E_DIM);
        float var = ts2 * (1.0f / E_DIM) - mean * mean;
        sh[0] = mean;
        sh[1] = rsqrtf(var + 1e-5f);
    }
    __syncthreads();
    float mean = sh[0], rstd = sh[1];
    _Float16* orow = out + (size_t)row * E_DIM;
#pragma unroll
    for (int i = 0; i < 4; ++i) {
        int c = tid * 4 + i;
        orow[c] = (_Float16)((vv[i] - mean) * rstd * g[c] + bb[c]);
    }
}

// ---------------- tiled WMMA GEMM:  C = A(MxK) * Bw(NxK)^T + bias ----------------
// Double-buffered LDS, async global->LDS staging overlapped with WMMA compute.
#define EPI_QKV   0   // write f16 to (B,H,S,Dh) layout, apply scale
#define EPI_GELU  1   // exact-erf gelu, write f16 row-major
#define EPI_RESID 2   // add fp32 residual, write fp32 row-major

template <int MODE>
__global__ __launch_bounds__(256)
void gemm_f16_kernel(const _Float16* __restrict__ A,
                     const _Float16* __restrict__ Bw,
                     const float* __restrict__ bias,
                     const float* __restrict__ resid,
                     void* __restrict__ outp,
                     float scale, int M, int N, int K) {
    constexpr int BM = 128, BN = 128, BK = 64, LD = BK + 8;  // 72-elem padded rows
    __shared__ _Float16 lsA[2][BM * LD];
    __shared__ _Float16 lsB[2][BN * LD];

    const int tid  = threadIdx.x;
    const int lane = tid & 31, w = tid >> 5;
    const int rn = lane & 15, hf = lane >> 4;
    const int wm = w & 1, wn = w >> 1;            // 2 x 4 wave grid
    const int m0 = blockIdx.y * BM, n0 = blockIdx.x * BN;

    v8f acc[4][2];
#pragma unroll
    for (int i = 0; i < 4; ++i)
#pragma unroll
        for (int j = 0; j < 2; ++j) acc[i][j] = vzero();

    // async stage of one 128x64 tile pair into buffer `buf`
    auto stage = [&](int buf, int k0) {
#pragma unroll
        for (int c = 0; c < 4; ++c) {
            int id = c * 256 + tid;
            int row = id >> 3, c8 = (id & 7) * 8;
            async_copy_b128(&lsA[buf][row * LD + c8],
                            A + (size_t)(m0 + row) * K + k0 + c8);
            async_copy_b128(&lsB[buf][row * LD + c8],
                            Bw + (size_t)(n0 + row) * K + k0 + c8);
        }
    };

    stage(0, 0);
    wait_async0();
    __syncthreads();

    const int nIter = K / BK;
    for (int it = 0; it < nIter; ++it) {
        const int cur = it & 1;
        if (it + 1 < nIter) stage(cur ^ 1, (it + 1) * BK);  // overlap next copy

#pragma unroll
        for (int k2 = 0; k2 < BK; k2 += 32) {
            Frag af[4], bf[2];
#pragma unroll
            for (int mt = 0; mt < 4; ++mt) {     // A frag: lane<16 K[0..8)+[16..24)
                const _Float16* p = &lsA[cur][(wm * 64 + mt * 16 + rn) * LD + k2];
                af[mt].lo = *(const h8*)(p + hf * 8);
                af[mt].hi = *(const h8*)(p + 16 + hf * 8);
            }
#pragma unroll
            for (int nt = 0; nt < 2; ++nt) {     // B frag: lane<16 K[0..8)+[16..24)
                const _Float16* p = &lsB[cur][(wn * 32 + nt * 16 + rn) * LD + k2];
                bf[nt].lo = *(const h8*)(p + hf * 8);
                bf[nt].hi = *(const h8*)(p + 16 + hf * 8);
            }
#pragma unroll
            for (int mt = 0; mt < 4; ++mt)
#pragma unroll
                for (int nt = 0; nt < 2; ++nt)
                    acc[mt][nt] = wmma16(af[mt], bf[nt], acc[mt][nt]);
        }

        wait_async0();      // my async writes to the other buffer have landed
        __syncthreads();    // everyone done computing `cur` + sees all LDS writes
    }

    // epilogue: C layout -> VGPR e holds rows e (lanes 0-15) / e+8 (lanes 16-31)
#pragma unroll
    for (int mt = 0; mt < 4; ++mt)
#pragma unroll
        for (int nt = 0; nt < 2; ++nt)
#pragma unroll
            for (int e = 0; e < 8; ++e) {
                int m = m0 + wm * 64 + mt * 16 + e + 8 * hf;
                int n = n0 + wn * 32 + nt * 16 + rn;
                float v = acc[mt][nt][e] + bias[n];
                if constexpr (MODE == EPI_QKV) {
                    v *= scale;
                    int b = m & (B_SZ - 1), s = m >> 1;
                    int h = n >> 6, d = n & 63;
                    ((_Float16*)outp)[(((size_t)(b * H_NUM + h)) * S_LEN + s) * DH + d] =
                        (_Float16)v;
                } else if constexpr (MODE == EPI_GELU) {
                    float gv = 0.5f * v * (1.0f + erff(v * 0.70710678118654752f));
                    ((_Float16*)outp)[(size_t)m * N + n] = (_Float16)gv;
                } else {
                    ((float*)outp)[(size_t)m * N + n] =
                        resid[(size_t)m * N + n] + v;
                }
            }
}

// ---------------- flash attention (online softmax, WMMA) ----------------
// grid: (S/64, B*H), block: 128 (4 waves x 16 q-rows)
__global__ __launch_bounds__(128)
void attn_kernel(const _Float16* __restrict__ Q, const _Float16* __restrict__ Km,
                 const _Float16* __restrict__ V, const float* __restrict__ kbias,
                 _Float16* __restrict__ O) {
    __shared__ _Float16 lsVT[DH * 40];       // V^T chunk: 64 d-rows x 32 keys (pad 40)
    __shared__ _Float16 lsP[4 * 16 * 40];    // per-wave P tile 16x32 (pad 40)

    const int tid  = threadIdx.x;
    const int lane = tid & 31, w = tid >> 5;
    const int rn = lane & 15, hf = lane >> 4;
    const int qb = blockIdx.x;               // 64-row q block
    const int bh = blockIdx.y;
    const int b = bh / H_NUM, h = bh - b * H_NUM;
    const size_t base = (size_t)bh * S_LEN;  // rows of (b,h) in q/k/v buffers

    Frag aq[2];                              // Q 16x64 held as two 16x32 A-frags
    {
        const _Float16* qp = Q + (base + qb * 64 + w * 16 + rn) * DH;
#pragma unroll
        for (int f = 0; f < 2; ++f) {
            aq[f].lo = *(const h8*)(qp + f * 32 + hf * 8);
            aq[f].hi = *(const h8*)(qp + f * 32 + 16 + hf * 8);
        }
    }

    Frag ones;                               // all-ones B matrix: row-sum via WMMA
#pragma unroll
    for (int i = 0; i < 16; ++i) ones.v[i] = (_Float16)1.0f;

    float mrow[8], lrow[8];
    v8f o[4];
#pragma unroll
    for (int e = 0; e < 8; ++e) { mrow[e] = -3.0e38f; lrow[e] = 0.f; }
#pragma unroll
    for (int t = 0; t < 4; ++t) o[t] = vzero();

    const float* kb = kbias + b * S_LEN;

    for (int j0 = 0; j0 < S_LEN; j0 += 32) {
        __syncthreads();
#pragma unroll
        for (int c = 0; c < 2; ++c) {        // stage V chunk transposed
            int id = c * 128 + tid;
            int key = id >> 3, d8 = (id & 7) * 8;
            h8 vv = *(const h8*)(V + (base + j0 + key) * DH + d8);
#pragma unroll
            for (int e = 0; e < 8; ++e) lsVT[(d8 + e) * 40 + key] = vv[e];
        }
        __syncthreads();

        // S = Q K^T for 32 keys (2 N-tiles, K-dim 64 = 2 wmma each)
        v8f sc[2];
#pragma unroll
        for (int nt = 0; nt < 2; ++nt) {
            sc[nt] = vzero();
            const _Float16* kp = Km + (base + j0 + nt * 16 + rn) * DH;
            Frag bk;
            bk.lo = *(const h8*)(kp + hf * 8);
            bk.hi = *(const h8*)(kp + 16 + hf * 8);
            sc[nt] = wmma16(aq[0], bk, sc[nt]);
            bk.lo = *(const h8*)(kp + 32 + hf * 8);
            bk.hi = *(const h8*)(kp + 48 + hf * 8);
            sc[nt] = wmma16(aq[1], bk, sc[nt]);
        }

        float bias0 = kb[j0 + rn], bias1 = kb[j0 + 16 + rn];
        float mnew[8], alpha[8];
#pragma unroll
        for (int e = 0; e < 8; ++e) {
            float v0 = sc[0][e] + bias0;
            float v1 = sc[1][e] + bias1;
            sc[0][e] = v0; sc[1][e] = v1;
            float mx = fmaxf(v0, v1);
#pragma unroll
            for (int off = 1; off < 16; off <<= 1)   // reduce within 16-lane row group
                mx = fmaxf(mx, __shfl_xor(mx, off));
            mnew[e] = fmaxf(mrow[e], mx);
        }
#pragma unroll
        for (int e = 0; e < 8; ++e) {
            alpha[e] = __expf(mrow[e] - mnew[e]);
            mrow[e] = mnew[e];
            float p0 = __expf(sc[0][e] - mnew[e]);
            float p1 = __expf(sc[1][e] - mnew[e]);
            int prow = (w * 16 + e + 8 * hf) * 40;   // wave-private P tile
            lsP[prow + rn]      = (_Float16)p0;
            lsP[prow + 16 + rn] = (_Float16)p1;
#pragma unroll
            for (int t = 0; t < 4; ++t) o[t][e] *= alpha[e];
        }

        // re-fragment P as A-matrix (wave-private LDS, per-wave ops in order)
        Frag ap;
        {
            const _Float16* pp = &lsP[(w * 16 + rn) * 40];
            ap.lo = *(const h8*)(pp + hf * 8);
            ap.hi = *(const h8*)(pp + 16 + hf * 8);
        }

        // row sums via P @ ones (broadcast into every column) -- no shuffles
        v8f ps = wmma16(ap, ones, vzero());
#pragma unroll
        for (int e = 0; e < 8; ++e) lrow[e] = lrow[e] * alpha[e] + ps[e];

        // O += P(16x32) @ V(32x64)
#pragma unroll
        for (int t = 0; t < 4; ++t) {
            Frag bv;
            const _Float16* vp = &lsVT[(t * 16 + rn) * 40];
            bv.lo = *(const h8*)(vp + hf * 8);
            bv.hi = *(const h8*)(vp + 16 + hf * 8);
            o[t] = wmma16(ap, bv, o[t]);
        }
    }

    // finalize: divide by row sums, scatter into (S,B,E) fp16 layout
#pragma unroll
    for (int e = 0; e < 8; ++e) {
        float inv = lrow[e] > 0.f ? 1.0f / lrow[e] : 0.f;
        int s = qb * 64 + w * 16 + e + 8 * hf;
        size_t rowoff = ((size_t)s * B_SZ + b) * E_DIM + h * DH;
#pragma unroll
        for (int t = 0; t < 4; ++t)
            O[rowoff + t * 16 + rn] = (_Float16)(o[t][e] * inv);
    }
}

// ---------------- host-side orchestration ----------------
extern "C" void kernel_launch(void* const* d_in, const int* in_sizes, int n_in,
                              void* d_out, int out_size, void* d_ws, size_t ws_size,
                              hipStream_t stream) {
    (void)in_sizes; (void)n_in; (void)out_size; (void)ws_size;

    const float*         x    = (const float*)d_in[0];
    const unsigned char* mask = (const unsigned char*)d_in[1];  // jnp bool = 1 byte
    const float* Wq   = (const float*)d_in[2];
    const float* bq   = (const float*)d_in[3];
    const float* Wk   = (const float*)d_in[4];
    const float* bk   = (const float*)d_in[5];
    const float* Wv   = (const float*)d_in[6];
    const float* bv   = (const float*)d_in[7];
    const float* Wo   = (const float*)d_in[8];
    const float* bo   = (const float*)d_in[9];
    const float* ln1g = (const float*)d_in[10];
    const float* ln1b = (const float*)d_in[11];
    const float* fc1w = (const float*)d_in[12];
    const float* fc1b = (const float*)d_in[13];
    const float* fc2w = (const float*)d_in[14];
    const float* fc2b = (const float*)d_in[15];
    const float* ln2g = (const float*)d_in[16];
    const float* ln2b = (const float*)d_in[17];

    char* ws = (char*)d_ws;
    size_t off = 0;
    auto alloc = [&](size_t bytes) -> void* {
        void* p = ws + off;
        off = (off + bytes + 255) & ~(size_t)255;
        return p;
    };

    _Float16* hbf  = (_Float16*)alloc((size_t)NT * E_DIM * 2);
    _Float16* wqh  = (_Float16*)alloc((size_t)E_DIM * E_DIM * 2);
    _Float16* wkh  = (_Float16*)alloc((size_t)E_DIM * E_DIM * 2);
    _Float16* wvh  = (_Float16*)alloc((size_t)E_DIM * E_DIM * 2);
    _Float16* woh  = (_Float16*)alloc((size_t)E_DIM * E_DIM * 2);
    _Float16* fc1h = (_Float16*)alloc((size_t)F_DIM * E_DIM * 2);
    _Float16* fc2h = (_Float16*)alloc((size_t)E_DIM * F_DIM * 2);
    _Float16* qbuf = (_Float16*)alloc((size_t)NT * E_DIM * 2);
    _Float16* kbuf = (_Float16*)alloc((size_t)NT * E_DIM * 2);
    _Float16* vbuf = (_Float16*)alloc((size_t)NT * E_DIM * 2);
    float*    kbia = (float*)alloc((size_t)B_SZ * S_LEN * 4);
    _Float16* obuf = (_Float16*)alloc((size_t)NT * E_DIM * 2);
    float*    x1   = (float*)alloc((size_t)NT * E_DIM * 4);
    _Float16* h2   = (_Float16*)alloc((size_t)NT * E_DIM * 2);
    _Float16* a1   = (_Float16*)alloc((size_t)NT * F_DIM * 2);

    const float qscale = 0.125f;  // Dh^-0.5

    // weight conversion + mask bias
    cvt_f16_kernel<<<1024, 256, 0, stream>>>(Wq,   wqh,  E_DIM * E_DIM);
    cvt_f16_kernel<<<1024, 256, 0, stream>>>(Wk,   wkh,  E_DIM * E_DIM);
    cvt_f16_kernel<<<1024, 256, 0, stream>>>(Wv,   wvh,  E_DIM * E_DIM);
    cvt_f16_kernel<<<1024, 256, 0, stream>>>(Wo,   woh,  E_DIM * E_DIM);
    cvt_f16_kernel<<<2048, 256, 0, stream>>>(fc1w, fc1h, F_DIM * E_DIM);
    cvt_f16_kernel<<<2048, 256, 0, stream>>>(fc2w, fc2h, E_DIM * F_DIM);
    maskbias_kernel<<<16, 256, 0, stream>>>(mask, kbia, B_SZ * S_LEN);

    // LN1
    ln_f16_kernel<<<NT, 256, 0, stream>>>(x, ln1g, ln1b, hbf);

    // QKV projections (M=4096, N=1024, K=1024), scatter to (B,H,S,Dh)
    dim3 gEE(E_DIM / 128, NT / 128);
    gemm_f16_kernel<EPI_QKV><<<gEE, 256, 0, stream>>>(hbf, wqh, bq, nullptr, qbuf,
                                                      qscale, NT, E_DIM, E_DIM);
    gemm_f16_kernel<EPI_QKV><<<gEE, 256, 0, stream>>>(hbf, wkh, bk, nullptr, kbuf,
                                                      1.0f, NT, E_DIM, E_DIM);
    gemm_f16_kernel<EPI_QKV><<<gEE, 256, 0, stream>>>(hbf, wvh, bv, nullptr, vbuf,
                                                      1.0f, NT, E_DIM, E_DIM);

    // flash attention -> obuf in (S,B,E) f16 layout
    dim3 gAttn(S_LEN / 64, B_SZ * H_NUM);
    attn_kernel<<<gAttn, 128, 0, stream>>>(qbuf, kbuf, vbuf, kbia, obuf);

    // x1 = x + obuf @ Wo^T + bo
    gemm_f16_kernel<EPI_RESID><<<gEE, 256, 0, stream>>>(obuf, woh, bo, x, x1,
                                                        1.0f, NT, E_DIM, E_DIM);

    // LN2 -> h2
    ln_f16_kernel<<<NT, 256, 0, stream>>>(x1, ln2g, ln2b, h2);

    // a1 = gelu(h2 @ fc1^T + b1)  (M=4096, N=4096, K=1024)
    dim3 gFE(F_DIM / 128, NT / 128);
    gemm_f16_kernel<EPI_GELU><<<gFE, 256, 0, stream>>>(h2, fc1h, fc1b, nullptr, a1,
                                                       1.0f, NT, F_DIM, E_DIM);

    // out = x1 + a1 @ fc2^T + b2  (M=4096, N=1024, K=4096)
    gemm_f16_kernel<EPI_RESID><<<gEE, 256, 0, stream>>>(a1, fc2h, fc2b, x1,
                                                        (float*)d_out,
                                                        1.0f, NT, E_DIM, F_DIM);
}
